// HourglassEncoder_23252952941092
// MI455X (gfx1250) — matrix operand order repaired and checked by reading, not compiled, not run on hardware
//
#include <hip/hip_runtime.h>
#include <math.h>

// ---------------------------------------------------------------------------
// Hourglass encoder for MI455X (gfx1250, wave32, WMMA).
// GEMMs are compute-bound (hundreds of FLOP/byte) -> bf16 WMMA 16x16x32 with
// fp32 accumulation, 128x128 block tiles, 32x64 wave tiles (8 WMMA / wave /
// k-step), software-pipelined global->LDS staging.
// Attention is fused flash-style (QK^T + online softmax + PV, all WMMA) to
// avoid materializing 2048x2048 scores (537 MB/layer would be HBM-bound).
// ---------------------------------------------------------------------------

#define NHEADS 8
#define DHEAD  64
#define INNER  512            // NHEADS * DHEAD
#define ATTN_SCALE 0.125f     // DHEAD ** -0.5

typedef __attribute__((ext_vector_type(16))) __bf16 v16bf;
typedef __attribute__((ext_vector_type(8)))  float  v8f;

union FragBF { v16bf v; uint4 q4[2]; };
union U4     { uint4 q; unsigned short s[8]; };

__device__ __forceinline__ unsigned short f2bf(float f) {
  unsigned u = __float_as_uint(f);
  u += 0x7fffu + ((u >> 16) & 1u);          // round-to-nearest-even
  return (unsigned short)(u >> 16);
}
__device__ __forceinline__ float bf2f(unsigned short h) {
  return __uint_as_float(((unsigned)h) << 16);
}
__device__ __forceinline__ float gelu_exact(float x) {
  return 0.5f * x * (1.0f + erff(x * 0.70710678f));
}
__device__ __forceinline__ v8f wmma_bf16(const FragBF& a, const FragBF& b, v8f c) {
  return __builtin_amdgcn_wmma_f32_16x16x32_bf16(false, a.v, false, b.v,
                                                 (short)0, c, false, false);
}

// ---------------------------------------------------------------------------
// GEMM: C[MxN] = A[MxK](bf16) * B[KxN](bf16)  (+bias)(+gelu)(+residual)
// Block = 256 threads (8 waves). Block tile 128x128, wave grid 4x2, wave tile
// 32x64 = 2x4 WMMA tiles, K step 32. Global loads for tile k+1 are issued
// right after the barrier so VMEM latency hides behind the 8 WMMAs of tile k.
// All M are multiples of 128, all N multiples of 128, all K multiples of 32
// for every GEMM in this model -> no bounds checks.
// A-frag (16x32, 16-bit): lane m=l&15; K = mh*8..+7 then 16+mh*8..+7 (mh=l>>4).
// B-frag (32x16): lane n=l&15; per-lane K contiguous -> B kept transposed in
// LDS so each fragment is two b128 LDS reads.
// ---------------------------------------------------------------------------
__global__ __launch_bounds__(256)
void gemm_bf16_wmma(const unsigned short* __restrict__ A,
                    const unsigned short* __restrict__ B,
                    const float* __restrict__ bias,
                    const float* __restrict__ res,
                    float* __restrict__ outF,
                    unsigned short* __restrict__ outH,
                    int M, int N, int K, int epi, int has_res, int out_bf) {
  __shared__ __align__(16) unsigned short Al[128 * 32];
  __shared__ __align__(16) unsigned short Bt[128 * 32];   // B transposed [n][k]
  const int tid = threadIdx.x, lane = tid & 31, wave = tid >> 5;
  const int wm = wave >> 1, wn = wave & 1;
  const int ml = lane & 15, mh = lane >> 4;
  const int bm = blockIdx.y * 128, bn = blockIdx.x * 128;

  v8f c[2][4];
  for (int i = 0; i < 2; i++)
    for (int j = 0; j < 4; j++) {
      v8f z = {0.f, 0.f, 0.f, 0.f, 0.f, 0.f, 0.f, 0.f};
      c[i][j] = z;
    }

  const int arow = tid >> 1, aseg = (tid & 1) * 16;   // A tile: 128 rows x 32
  const int krow = tid >> 3, nseg = (tid & 7) * 16;   // B tile: 32 rows x 128

  uint4 a0, a1;      // in-flight A tile (16 bf16 / thread)
  U4 b0, b1;         // in-flight B tile (16 bf16 / thread)
  {
    const uint4* gA = (const uint4*)(A + (size_t)(bm + arow) * K + aseg);
    a0 = gA[0]; a1 = gA[1];
    const uint4* gB = (const uint4*)(B + (size_t)krow * N + bn + nseg);
    b0.q = gB[0]; b1.q = gB[1];
  }

  for (int k0 = 0; k0 < K; k0 += 32) {
    // ---- commit in-flight tile to LDS (B transposed) ----
    *(uint4*)&Al[arow * 32 + aseg]     = a0;
    *(uint4*)&Al[arow * 32 + aseg + 8] = a1;
#pragma unroll
    for (int j = 0; j < 8; j++) {
      Bt[(nseg + j) * 32 + krow]     = b0.s[j];
      Bt[(nseg + 8 + j) * 32 + krow] = b1.s[j];
    }
    __syncthreads();

    // ---- issue next tile's global loads (hidden behind WMMAs below) ----
    if (k0 + 32 < K) {
      const int kn = k0 + 32;
      const uint4* gA = (const uint4*)(A + (size_t)(bm + arow) * K + kn + aseg);
      a0 = gA[0]; a1 = gA[1];
      const uint4* gB = (const uint4*)(B + (size_t)(kn + krow) * N + bn + nseg);
      b0.q = gB[0]; b1.q = gB[1];
      __builtin_prefetch(A + (size_t)(bm + arow) * K + kn + 32 + aseg, 0, 1);
      __builtin_prefetch(B + (size_t)(kn + 32 + krow) * N + bn + nseg, 0, 1);
    }

    // ---- fragments + 8 WMMA ----
    FragBF af[2], bfr[4];
#pragma unroll
    for (int i = 0; i < 2; i++) {
      const int row = wm * 32 + i * 16 + ml;
      af[i].q4[0] = *(const uint4*)&Al[row * 32 + mh * 8];
      af[i].q4[1] = *(const uint4*)&Al[row * 32 + 16 + mh * 8];
    }
#pragma unroll
    for (int j = 0; j < 4; j++) {
      const int col = wn * 64 + j * 16 + ml;
      bfr[j].q4[0] = *(const uint4*)&Bt[col * 32 + mh * 8];
      bfr[j].q4[1] = *(const uint4*)&Bt[col * 32 + 16 + mh * 8];
    }
#pragma unroll
    for (int i = 0; i < 2; i++)
#pragma unroll
      for (int j = 0; j < 4; j++)
        c[i][j] = wmma_bf16(af[i], bfr[j], c[i][j]);
    __syncthreads();
  }

#pragma unroll
  for (int i = 0; i < 2; i++)
#pragma unroll
    for (int j = 0; j < 4; j++)
#pragma unroll
      for (int r = 0; r < 8; r++) {
        const int row = bm + wm * 32 + i * 16 + r + mh * 8;
        const int col = bn + wn * 64 + j * 16 + ml;
        float v = c[i][j][r];
        if (epi >= 1) v += bias[col];
        if (epi == 2) v = gelu_exact(v);
        if (has_res) v += res[(size_t)row * N + col];
        if (out_bf) outH[(size_t)row * N + col] = f2bf(v);
        else        outF[(size_t)row * N + col] = v;
      }
}

// ---------------------------------------------------------------------------
// Flash attention. Grid = (n/64, b*H). Block = 128 threads (4 waves), each
// wave owns 16 query rows. Keys streamed in chunks of 32: 4 WMMA for QK^T,
// online softmax via shfl_xor row reductions (rows of a 16x16 C tile live
// across 16 lanes of one half-wave), P restaged through LDS into A layout,
// 4 WMMA for P@V. K/V staging for chunk kc+1 is issued while chunk kc's
// WMMAs run. Q: (b*n, INNER) bf16. KV: (b*n, 2*INNER), K then V per head.
// ---------------------------------------------------------------------------
__global__ __launch_bounds__(128)
void flash_attn(const unsigned short* __restrict__ Q,
                const unsigned short* __restrict__ KV,
                const unsigned char* __restrict__ mask,
                unsigned short* __restrict__ O,
                int n, float scale) {
  __shared__ __align__(16) unsigned short Kl[32 * 64];    // [key][d]
  __shared__ __align__(16) unsigned short Vt[64 * 32];    // [d][key]
  __shared__ __align__(16) unsigned short Pl[4][16 * 32]; // per-wave P tile
  const int b = blockIdx.y / NHEADS, h = blockIdx.y % NHEADS;
  const int tid = threadIdx.x, lane = tid & 31, wave = tid >> 5;
  const int ml = lane & 15, mh = lane >> 4;
  const int qrow = blockIdx.x * 64 + wave * 16 + ml;

  const size_t qoff = ((size_t)b * n + qrow) * INNER + h * DHEAD + mh * 8;
  FragBF qf[2];
  qf[0].q4[0] = *(const uint4*)&Q[qoff];
  qf[0].q4[1] = *(const uint4*)&Q[qoff + 16];
  qf[1].q4[0] = *(const uint4*)&Q[qoff + 32];
  qf[1].q4[1] = *(const uint4*)&Q[qoff + 48];

  v8f o[4];
  for (int t = 0; t < 4; t++) {
    v8f z = {0.f, 0.f, 0.f, 0.f, 0.f, 0.f, 0.f, 0.f};
    o[t] = z;
  }
  float mrun[8], lrun[8];
  for (int r = 0; r < 8; r++) { mrun[r] = -3.0e38f; lrun[r] = 0.f; }

  const int skey = tid >> 2;          // 0..31: key row this thread stages
  const int sseg = (tid & 3) * 16;    // 16 of 64 head dims

  uint4 kA, kB;   // in-flight K chunk
  U4 vA, vB;      // in-flight V chunk
  {
    const size_t kvrow = ((size_t)b * n + skey) * (2 * INNER) + h * DHEAD;
    kA = *(const uint4*)&KV[kvrow + sseg];
    kB = *(const uint4*)&KV[kvrow + sseg + 8];
    vA.q = *(const uint4*)&KV[kvrow + INNER + sseg];
    vB.q = *(const uint4*)&KV[kvrow + INNER + sseg + 8];
  }

  for (int kc = 0; kc < n; kc += 32) {
    // ---- commit in-flight K (as-is) and V (transposed) to LDS ----
    *(uint4*)&Kl[skey * 64 + sseg]     = kA;
    *(uint4*)&Kl[skey * 64 + sseg + 8] = kB;
#pragma unroll
    for (int j = 0; j < 8; j++) {
      Vt[(sseg + j) * 32 + skey]     = vA.s[j];
      Vt[(sseg + j + 8) * 32 + skey] = vB.s[j];
    }
    __syncthreads();

    // ---- issue next chunk's loads (overlap with WMMA/softmax below) ----
    if (kc + 32 < n) {
      const size_t kvrow = ((size_t)b * n + kc + 32 + skey) * (2 * INNER) + h * DHEAD;
      kA = *(const uint4*)&KV[kvrow + sseg];
      kB = *(const uint4*)&KV[kvrow + sseg + 8];
      vA.q = *(const uint4*)&KV[kvrow + INNER + sseg];
      vB.q = *(const uint4*)&KV[kvrow + INNER + sseg + 8];
    }

    // ---- S = Q K^T (16 rows x 32 keys): 2 d-steps x 2 key tiles ----
    v8f s0, s1;
    {
      v8f z = {0.f, 0.f, 0.f, 0.f, 0.f, 0.f, 0.f, 0.f};
      s0 = z; s1 = z;
    }
#pragma unroll
    for (int ds = 0; ds < 2; ds++) {
      FragBF k0f, k1f;   // B frags: d = mh*8..+7 then 16+mh*8..+7 inside K rows
      k0f.q4[0] = *(const uint4*)&Kl[ml * 64 + ds * 32 + mh * 8];
      k0f.q4[1] = *(const uint4*)&Kl[ml * 64 + ds * 32 + 16 + mh * 8];
      k1f.q4[0] = *(const uint4*)&Kl[(16 + ml) * 64 + ds * 32 + mh * 8];
      k1f.q4[1] = *(const uint4*)&Kl[(16 + ml) * 64 + ds * 32 + 16 + mh * 8];
      s0 = wmma_bf16(qf[ds], k0f, s0);
      s1 = wmma_bf16(qf[ds], k1f, s1);
    }

    // ---- online softmax update; write P into LDS in C layout ----
    bool m0 = true, m1 = true;
    if (mask) {
      m0 = mask[(size_t)b * n + kc + ml] != 0;
      m1 = mask[(size_t)b * n + kc + 16 + ml] != 0;
    }
#pragma unroll
    for (int r = 0; r < 8; r++) {
      float a0 = m0 ? s0[r] * scale : -3.0e38f;
      float a1 = m1 ? s1[r] * scale : -3.0e38f;
      float cm = fmaxf(a0, a1);
      for (int off = 1; off < 16; off <<= 1) cm = fmaxf(cm, __shfl_xor(cm, off, 32));
      const float nm = fmaxf(mrun[r], cm);
      const float al = __expf(mrun[r] - nm);
      const float p0 = __expf(a0 - nm);
      const float p1 = __expf(a1 - nm);
      float rs = p0 + p1;
      for (int off = 1; off < 16; off <<= 1) rs += __shfl_xor(rs, off, 32);
      lrun[r] = lrun[r] * al + rs;
      mrun[r] = nm;
      o[0][r] *= al; o[1][r] *= al; o[2][r] *= al; o[3][r] *= al;
      const int prow = r + 8 * mh;
      Pl[wave][prow * 32 + ml]      = f2bf(p0);
      Pl[wave][prow * 32 + 16 + ml] = f2bf(p1);
    }
    __syncthreads();

    // ---- O += P @ V ----
    FragBF pf;
    pf.q4[0] = *(const uint4*)&Pl[wave][ml * 32 + mh * 8];
    pf.q4[1] = *(const uint4*)&Pl[wave][ml * 32 + 16 + mh * 8];
#pragma unroll
    for (int t = 0; t < 4; t++) {
      FragBF vf;
      const int dcol = t * 16 + ml;
      vf.q4[0] = *(const uint4*)&Vt[dcol * 32 + mh * 8];
      vf.q4[1] = *(const uint4*)&Vt[dcol * 32 + 16 + mh * 8];
      o[t] = wmma_bf16(pf, vf, o[t]);
    }
    __syncthreads();
  }

#pragma unroll
  for (int r = 0; r < 8; r++) {
    const float inv = lrun[r] > 0.f ? 1.0f / lrun[r] : 0.f;
    const int row = blockIdx.x * 64 + wave * 16 + r + 8 * mh;
    const size_t obase = ((size_t)b * n + row) * INNER + h * DHEAD;
#pragma unroll
    for (int t = 0; t < 4; t++)
      O[obase + t * 16 + ml] = f2bf(o[t][r] * inv);
  }
}

// ---------------------------------------------------------------------------
// Valley attention: 1 query attends to its 2 source tokens. Memory-bound;
// one wave per (token, head), two head-dim elements per lane.
// ---------------------------------------------------------------------------
__global__ __launch_bounds__(256)
void valley_attn(const unsigned short* __restrict__ Qb,
                 const unsigned short* __restrict__ KVb,
                 const unsigned char* __restrict__ mask,
                 unsigned short* __restrict__ Ob,
                 int rows, float scale) {
  const int gid = blockIdx.x * 8 + (threadIdx.x >> 5);
  if (gid >= rows * NHEADS) return;
  const int lane = threadIdx.x & 31;
  const int tok = gid / NHEADS, h = gid % NHEADS;
  const size_t qo = (size_t)tok * INNER + h * DHEAD;
  const size_t c0 = ((size_t)tok * 2) * (2 * INNER) + h * DHEAD;
  const size_t c1 = ((size_t)tok * 2 + 1) * (2 * INNER) + h * DHEAD;
  const float q0 = bf2f(Qb[qo + lane]), q1 = bf2f(Qb[qo + 32 + lane]);
  float d0 = q0 * bf2f(KVb[c0 + lane]) + q1 * bf2f(KVb[c0 + 32 + lane]);
  float d1 = q0 * bf2f(KVb[c1 + lane]) + q1 * bf2f(KVb[c1 + 32 + lane]);
  for (int off = 16; off > 0; off >>= 1) {
    d0 += __shfl_xor(d0, off, 32);
    d1 += __shfl_xor(d1, off, 32);
  }
  float s0v = d0 * scale, s1v = d1 * scale;
  if (mask) {
    if (!mask[(size_t)tok * 2])     s0v = -3.0e38f;
    if (!mask[(size_t)tok * 2 + 1]) s1v = -3.0e38f;
  }
  const float mx = fmaxf(s0v, s1v);
  float p0 = __expf(s0v - mx), p1 = __expf(s1v - mx);
  const float inv = 1.0f / (p0 + p1);
  p0 *= inv; p1 *= inv;
  const float v0a = bf2f(KVb[c0 + INNER + lane]);
  const float v1a = bf2f(KVb[c1 + INNER + lane]);
  const float v0b = bf2f(KVb[c0 + INNER + 32 + lane]);
  const float v1b = bf2f(KVb[c1 + INNER + 32 + lane]);
  Ob[qo + lane]      = f2bf(p0 * v0a + p1 * v1a);
  Ob[qo + 32 + lane] = f2bf(p0 * v0b + p1 * v1b);
}

// ---------------------------------------------------------------------------
// LayerNorm: fp32 in, bf16 out. One row per 128-thread block (4 waves).
// ---------------------------------------------------------------------------
__global__ __launch_bounds__(128)
void layernorm_bf16(const float* __restrict__ x, const float* __restrict__ g,
                    const float* __restrict__ be, unsigned short* __restrict__ out,
                    int d) {
  const int row = blockIdx.x;
  const float* xr = x + (size_t)row * d;
  float s = 0.f, s2 = 0.f;
  for (int i = threadIdx.x; i < d; i += 128) {
    const float v = xr[i];
    s += v; s2 += v * v;
  }
  for (int off = 16; off > 0; off >>= 1) {
    s += __shfl_xor(s, off, 32);
    s2 += __shfl_xor(s2, off, 32);
  }
  __shared__ float sm[8];
  const int w = threadIdx.x >> 5, l = threadIdx.x & 31;
  if (l == 0) { sm[w] = s; sm[4 + w] = s2; }
  __syncthreads();
  s  = sm[0] + sm[1] + sm[2] + sm[3];
  s2 = sm[4] + sm[5] + sm[6] + sm[7];
  const float mu = s / d;
  const float var = s2 / d - mu * mu;
  const float rs = rsqrtf(var + 1e-5f);
  unsigned short* orow = out + (size_t)row * d;
  for (int i = threadIdx.x; i < d; i += 128)
    orow[i] = f2bf((xr[i] - mu) * rs * g[i] + be[i]);
}

__global__ void cvt_f32_bf16(const float* __restrict__ in,
                             unsigned short* __restrict__ out, long long n) {
  const long long i = (long long)blockIdx.x * blockDim.x + threadIdx.x;
  if (i < n) out[i] = f2bf(in[i]);
}

__global__ void down_mean(const float* __restrict__ x, float* __restrict__ out,
                          long long total, int d) {
  const long long i = (long long)blockIdx.x * blockDim.x + threadIdx.x;
  if (i >= total) return;
  const long long r = i / d;
  const int c = (int)(i % d);
  out[i] = 0.5f * (x[(2 * r) * (long long)d + c] + x[(2 * r + 1) * (long long)d + c]);
}

__global__ void mask_down(const unsigned char* __restrict__ in,
                          unsigned char* __restrict__ out, int n) {
  const int i = blockIdx.x * blockDim.x + threadIdx.x;
  if (i < n) out[i] = (in[2 * i] | in[2 * i + 1]) ? 1 : 0;
}

__global__ void mask_to_float(const unsigned char* __restrict__ in,
                              float* __restrict__ out, int n) {
  const int i = blockIdx.x * blockDim.x + threadIdx.x;
  if (i < n) out[i] = in[i] ? 1.0f : 0.0f;
}

// ---------------------------------------------------------------------------
// Host orchestration.
// Input order: x, mask, then params flattened in dict/list order:
//   per level: 4 pre layers + 1 valley layer, each
//   (ln1_g, ln1_b, wq, wkv, wo, bo, ln2_g, ln2_b, w1, b1, w2, b2),
//   then dp_ln_g, dp_ln_b, dp_w, dp_b. Level1 d=512,hid=2048; Level2 d=256.
// ---------------------------------------------------------------------------
extern "C" void kernel_launch(void* const* d_in, const int* in_sizes, int n_in,
                              void* d_out, int out_size, void* d_ws, size_t ws_size,
                              hipStream_t stream) {
  (void)in_sizes; (void)n_in; (void)out_size; (void)ws_size;
  const float* x_in = (const float*)d_in[0];
  const unsigned char* mask_in = (const unsigned char*)d_in[1];
  int cur = 2;
  auto nf = [&]() -> const float* { return (const float*)d_in[cur++]; };

  char* wp = (char*)d_ws;
  auto alloc = [&](size_t bytes) -> void* {
    void* r = (void*)wp;
    wp += (bytes + 255) & ~(size_t)255;
    return r;
  };

  auto cvt = [&](const float* src, unsigned short* dst, long long nels) {
    cvt_f32_bf16<<<dim3((unsigned)((nels + 255) / 256)), dim3(256), 0, stream>>>(src, dst, nels);
  };

  struct LayerPtrs { const float* f[12]; unsigned short* h[5]; };
  auto load_layer = [&](LayerPtrs& L, int d, int hid) {
    for (int i = 0; i < 12; i++) L.f[i] = nf();
    const int fidx[5] = {2, 3, 4, 8, 10};   // wq, wkv, wo, w1, w2
    const size_t sz[5] = {(size_t)d * INNER, (size_t)d * 2 * INNER,
                          (size_t)INNER * d, (size_t)d * hid, (size_t)hid * d};
    for (int i = 0; i < 5; i++) {
      L.h[i] = (unsigned short*)alloc(sz[i] * 2);
      cvt(L.f[fidx[i]], L.h[i], (long long)sz[i]);
    }
  };

  auto gemm = [&](const unsigned short* A, const unsigned short* B, const float* bias,
                  const float* res, float* oF, unsigned short* oH,
                  int M, int N, int K, int epi, int hres, int obf) {
    gemm_bf16_wmma<<<dim3(N / 128, M / 128), dim3(256), 0, stream>>>(
        A, B, bias, res, oF, oH, M, N, K, epi, hres, obf);
  };

  auto run_level = [&](const float* x_src, const unsigned char* m_src,
                       int bsz, int n, int d, int hid,
                       float** next_x, unsigned char** next_m, float* final_out) {
    const int R = bsz * n;
    const int mlen = n / 2;
    const int Rm = bsz * mlen;

    LayerPtrs pre[4], val;
    for (int i = 0; i < 4; i++) load_layer(pre[i], d, hid);
    load_layer(val, d, hid);
    const float* dp_ln_g = nf();
    const float* dp_ln_b = nf();
    const float* dp_w = nf();
    const float* dp_b = nf();
    unsigned short* dp_w_h = (unsigned short*)alloc((size_t)d * (d / 2) * 2);
    cvt(dp_w, dp_w_h, (long long)d * (d / 2));

    float* x = (float*)alloc((size_t)R * d * 4);
    hipMemcpyAsync(x, x_src, (size_t)R * d * 4, hipMemcpyDeviceToDevice, stream);
    unsigned short* xn   = (unsigned short*)alloc((size_t)R * d * 2);
    unsigned short* qb   = (unsigned short*)alloc((size_t)R * INNER * 2);
    unsigned short* kvb  = (unsigned short*)alloc((size_t)R * 2 * INNER * 2);
    unsigned short* attn = (unsigned short*)alloc((size_t)R * INNER * 2);
    unsigned short* h1   = (unsigned short*)alloc((size_t)R * hid * 2);
    unsigned short* ctx  = (unsigned short*)alloc((size_t)R * d * 2);
    float* down = (float*)alloc((size_t)Rm * d * 4);
    unsigned char* dm = (unsigned char*)alloc((size_t)Rm);

    // ---- pre transformer layers (self-attention) ----
    for (int li = 0; li < 4; li++) {
      LayerPtrs& L = pre[li];
      layernorm_bf16<<<R, 128, 0, stream>>>(x, L.f[0], L.f[1], xn, d);
      gemm(xn, L.h[0], nullptr, nullptr, nullptr, qb, R, INNER, d, 0, 0, 1);
      gemm(xn, L.h[1], nullptr, nullptr, nullptr, kvb, R, 2 * INNER, d, 0, 0, 1);
      flash_attn<<<dim3(n / 64, bsz * NHEADS), 128, 0, stream>>>(
          qb, kvb, m_src, attn, n, ATTN_SCALE);
      gemm(attn, L.h[2], L.f[5], x, x, nullptr, R, d, INNER, 1, 1, 0);
      layernorm_bf16<<<R, 128, 0, stream>>>(x, L.f[6], L.f[7], xn, d);
      gemm(xn, L.h[3], L.f[9], nullptr, nullptr, h1, R, hid, d, 2, 0, 1);
      gemm(h1, L.h[4], L.f[11], x, x, nullptr, R, d, hid, 1, 1, 0);
    }

    // ---- downsample (mean of s=2) + mask pool + ctx bf16 ----
    down_mean<<<dim3((unsigned)(((long long)Rm * d + 255) / 256)), 256, 0, stream>>>(
        x, down, (long long)Rm * d, d);
    mask_down<<<(Rm + 255) / 256, 256, 0, stream>>>(m_src, dm, Rm);
    cvt(x, ctx, (long long)R * d);

    // ---- valley layer: pooled token attends to its 2 source tokens ----
    layernorm_bf16<<<Rm, 128, 0, stream>>>(down, val.f[0], val.f[1], xn, d);
    gemm(xn, val.h[0], nullptr, nullptr, nullptr, qb, Rm, INNER, d, 0, 0, 1);
    gemm(ctx, val.h[1], nullptr, nullptr, nullptr, kvb, R, 2 * INNER, d, 0, 0, 1);
    valley_attn<<<(Rm * NHEADS + 7) / 8, 256, 0, stream>>>(
        qb, kvb, m_src, attn, Rm, ATTN_SCALE);
    gemm(attn, val.h[2], val.f[5], down, down, nullptr, Rm, d, INNER, 1, 1, 0);
    layernorm_bf16<<<Rm, 128, 0, stream>>>(down, val.f[6], val.f[7], xn, d);
    gemm(xn, val.h[3], val.f[9], nullptr, nullptr, h1, Rm, hid, d, 2, 0, 1);
    gemm(h1, val.h[4], val.f[11], down, down, nullptr, Rm, d, hid, 1, 1, 0);

    // ---- projection to d/2 ----
    layernorm_bf16<<<Rm, 128, 0, stream>>>(down, dp_ln_g, dp_ln_b, xn, d);
    float* xo = final_out ? final_out : (float*)alloc((size_t)Rm * (d / 2) * 4);
    gemm(xn, dp_w_h, dp_b, nullptr, xo, nullptr, Rm, d / 2, d, 1, 0, 0);
    *next_x = xo;
    *next_m = dm;
  };

  float* x2; unsigned char* m2;
  run_level(x_in, mask_in, 4, 2048, 512, 2048, &x2, &m2, nullptr);
  float* hout; unsigned char* m3;
  run_level(x2, m2, 4, 1024, 256, 1024, &hout, &m3, (float*)d_out);

  // outputs: h (4*512*128 fp32) followed by downsampled mask (4*512)
  mask_to_float<<<(2048 + 255) / 256, 256, 0, stream>>>(
      m3, (float*)d_out + 4 * 512 * 128, 2048);
}
